// Block_88562225643805
// MI455X (gfx1250) — matrix operand order repaired and checked by reading, not compiled, not run on hardware
//
#include <hip/hip_runtime.h>
#include <hip/hip_bf16.h>
#include <cstdint>

// ---------- types ----------
typedef __bf16 bf16_t;
typedef bf16_t v16bf __attribute__((ext_vector_type(16)));
typedef float  v8f   __attribute__((ext_vector_type(8)));
typedef unsigned short u16;
typedef unsigned int   u32;
typedef unsigned long long u64;
typedef u32 u32x4 __attribute__((ext_vector_type(4)));
typedef int i32x8 __attribute__((ext_vector_type(8)));
typedef int i32x4 __attribute__((ext_vector_type(4)));

union FragA { v16bf v; u32 w[8]; u16 u[16]; };
union AccF  { v8f v; float f[8]; };

__device__ __forceinline__ u16 f2bf(float x) {
  u32 u = __float_as_uint(x);
  return (u16)((u + 0x7FFFu + ((u >> 16) & 1u)) >> 16);   // RNE
}
// A-frag k order used here (contiguous, same as B): lane holds row m=lane&15; element pair v holds
// k = 2v + 16*(lane>>4) + {0,1}
__device__ __forceinline__ int a_kofs(int v) { return 2 * v; }

// ---------- Tensor Data Mover: 2D bf16 tile -> LDS (D# per ISA 08 §8.3/8.4) ----------
// This toolchain exposes the 6-arg builtin:
//   (uint32x4 g0, int32x8 g1, int32x4 g2, int32x4 g3, int32x8 pad, i32 cpol)
__device__ __forceinline__ void tdm_load_2d(
    u32 lds_addr, const void* gaddr, u32 tensor_d0, u32 tensor_d1,
    u32 tile_d0, u32 tile_d1, u64 stride0_elems)
{
  u64 ga = (u64)(uintptr_t)gaddr;
  u32x4 g0;
  g0[0] = 1u;                                              // count=1 (valid), user
  g0[1] = lds_addr;                                        // LDS byte address
  g0[2] = (u32)ga;                                         // global_addr[31:0]
  g0[3] = (u32)((ga >> 32) & 0x01FFFFFFu) | (2u << 30);    // ga[56:32] | type=2
  i32x8 g1;
  g1[0] = (int)(1u << 16);                                 // data_size=1 (2 bytes)
  g1[1] = (int)((tensor_d0 & 0xFFFFu) << 16);              // tensor_dim0[15:0]
  g1[2] = (int)((tensor_d0 >> 16) | ((tensor_d1 & 0xFFFFu) << 16));
  g1[3] = (int)((tensor_d1 >> 16) | (tile_d0 << 16));      // tile_dim0
  g1[4] = (int)(tile_d1 & 0xFFFFu);                        // tile_dim1 (tile_dim2=0)
  g1[5] = (int)(u32)(stride0_elems & 0xFFFFFFFFu);         // tensor_dim0_stride lo
  g1[6] = (int)(u32)((stride0_elems >> 32) & 0xFFFFu);     // stride hi (stride1=0)
  g1[7] = 0;
  i32x4 z4 = {0, 0, 0, 0};
  i32x8 z8 = {0, 0, 0, 0, 0, 0, 0, 0};
  __builtin_amdgcn_tensor_load_to_lds(g0, g1, z4, z4, z8, 0);
}

#define BM 128
#define BN 64
#define BK 32

// ---------- fused GEMM: out = epi(A_bf16[M,K] @ Wt_bf16[N,K]^T + bias [+resid]) ----------
// EPI 0: bf16 out;  EPI 1: bf16 out + exact GELU;  EPI 2: f32 out + residual
template<int EPI>
__global__ __launch_bounds__(256) void gemm_wmma(
    const u16* __restrict__ A, const u16* __restrict__ Wt,
    const float* __restrict__ bias, const float* __restrict__ resid,
    void* __restrict__ outp, int M, int N, int K)
{
  __shared__ __align__(16) u16 lA[2][BM * BK];   // 16 KB
  __shared__ __align__(16) u16 lB[2][BN * BK];   // 8 KB
  const int tid = threadIdx.x;
  const int wid = tid >> 5, lane = tid & 31;
  const int lm = lane & 15, hf = lane >> 4;
  const int m0 = blockIdx.y * BM, n0 = blockIdx.x * BN;

  AccF acc[4];
#pragma unroll
  for (int t = 0; t < 4; ++t) { v8f z = {0,0,0,0,0,0,0,0}; acc[t].v = z; }

  // prime the pipeline: wave 0 drives the TDM
  if (wid == 0) {
    tdm_load_2d((u32)(uintptr_t)&lA[0][0], A + (size_t)m0 * K, K, M, BK, BM, (u64)K);
    tdm_load_2d((u32)(uintptr_t)&lB[0][0], Wt + (size_t)n0 * K, K, N, BK, BN, (u64)K);
  }

  for (int k0 = 0; k0 < K; k0 += BK) {
    const int buf = (k0 >> 5) & 1;
    if (wid == 0) __builtin_amdgcn_s_wait_tensorcnt(0);
    __syncthreads();                         // tile[buf] landed; prev reads done
    if (wid == 0 && (k0 + BK) < K) {         // prefetch next tiles (overlaps WMMA)
      tdm_load_2d((u32)(uintptr_t)&lA[buf ^ 1][0],
                  A + (size_t)m0 * K + (k0 + BK), K, M, BK, BM, (u64)K);
      tdm_load_2d((u32)(uintptr_t)&lB[buf ^ 1][0],
                  Wt + (size_t)n0 * K + (k0 + BK), K, N, BK, BN, (u64)K);
    }

    FragA af;
    const u16* arow = &lA[buf][(wid * 16 + lm) * BK];
#pragma unroll
  for (int v = 0; v < 8; ++v) af.w[v] = *(const u32*)(arow + a_kofs(v) + 16 * hf);

#pragma unroll
    for (int t = 0; t < 4; ++t) {
      FragA bfr;   // B 32x16: lane col n=lane&15, k = 16*(lane>>4) + e (contiguous)
      const u16* brow = &lB[buf][(t * 16 + lm) * BK + 16 * hf];
#pragma unroll
      for (int v = 0; v < 8; ++v) bfr.w[v] = *(const u32*)(brow + 2 * v);
      acc[t].v = __builtin_amdgcn_wmma_f32_16x16x32_bf16(
          false, af.v, false, bfr.v, (short)0, acc[t].v, false, false);
    }
  }

  // epilogue: C layout lane n=lane&15, row m = r + 8*(lane>>4)
#pragma unroll
  for (int t = 0; t < 4; ++t) {
    int gn = n0 + t * 16 + lm;
    float bv = bias[gn];
#pragma unroll
    for (int r = 0; r < 8; ++r) {
      int gm = m0 + wid * 16 + r + 8 * hf;
      float val = acc[t].f[r] + bv;
      if (EPI == 1) val = 0.5f * val * (1.0f + erff(val * 0.70710678118f));
      if (EPI == 2) {
        ((float*)outp)[(size_t)gm * N + gn] = val + resid[(size_t)gm * N + gn];
      } else {
        ((u16*)outp)[(size_t)gm * N + gn] = f2bf(val);
      }
    }
  }
}

// ---------- one-shot weight convert: f32 [K,N] -> bf16 transposed [N,K] ----------
__global__ __launch_bounds__(256) void transpose_w_bf16(
    const float* __restrict__ W, u16* __restrict__ Wt, int K, int N)
{
  __shared__ float tile[32][33];
  int kb = blockIdx.y * 32, nb = blockIdx.x * 32;
  int tx = threadIdx.x & 31, ty = threadIdx.x >> 5;    // 8 row-strides
#pragma unroll
  for (int r = ty; r < 32; r += 8)
    tile[r][tx] = W[(size_t)(kb + r) * N + nb + tx];
  __syncthreads();
#pragma unroll
  for (int r = ty; r < 32; r += 8)
    Wt[(size_t)(nb + r) * K + kb + tx] = f2bf(tile[tx][r]);
}

// ---------- LayerNorm (f32 in -> bf16 out), one row per block ----------
#define CED 1024
__global__ __launch_bounds__(256) void layernorm_k(
    const float* __restrict__ x, const float* __restrict__ w,
    const float* __restrict__ b, u16* __restrict__ out)
{
  __shared__ float s1[256];
  __shared__ float s2[256];
  int row = blockIdx.x, tid = threadIdx.x;
  const float* xr = x + (size_t)row * CED;
  float v[4]; float sum = 0.f, sq = 0.f;
#pragma unroll
  for (int j = 0; j < 4; ++j) { v[j] = xr[tid + 256 * j]; sum += v[j]; sq += v[j] * v[j]; }
  s1[tid] = sum; s2[tid] = sq; __syncthreads();
  for (int st = 128; st > 0; st >>= 1) {
    if (tid < st) { s1[tid] += s1[tid + st]; s2[tid] += s2[tid + st]; }
    __syncthreads();
  }
  float mu = s1[0] * (1.0f / CED);
  float var = s2[0] * (1.0f / CED) - mu * mu;
  float rstd = rsqrtf(var + 1e-5f);
  u16* orow = out + (size_t)row * CED;
#pragma unroll
  for (int j = 0; j < 4; ++j) {
    int c = tid + 256 * j;
    orow[c] = f2bf((v[j] - mu) * rstd * w[c] + b[c]);
  }
}

// ---------- causal flash attention: qkv bf16 [B,T,3C] -> y bf16 [B,T,C] ----------
#define T_SEQ 2048
#define NH 16
#define C3 3072
__global__ __launch_bounds__(128) void attn_flash(
    const u16* __restrict__ qkv, u16* __restrict__ y)
{
  __shared__ __align__(16) u16 Pbuf[4][16 * 32];        // per-wave P scratch (4 KB)
  __shared__ __align__(16) u16 Vt[4][2][32 * 64];       // per-wave V tiles  (32 KB)
  const int wid = threadIdx.x >> 5, lane = threadIdx.x & 31;
  const int lm = lane & 15, hf = lane >> 4;
  int task = blockIdx.x * 4 + wid;
  const int QT = T_SEQ / 16;
  int qt = task % QT;
  int h  = (task / QT) % NH;
  int b  = task / (QT * NH);
  int q0 = qt * 16;

  const u16* base  = qkv + (size_t)b * T_SEQ * C3 + h * 64;
  const u16* vbase = base + 2 * CED;
  const int  nkv   = q0 + 16;                           // keys processed: [0, nkv)

  FragA qf[2];                                          // Q tile 16x64, two A-frags
  {
    const u16* qrow = base + (size_t)(q0 + lm) * C3;
#pragma unroll
    for (int kh = 0; kh < 2; ++kh)
#pragma unroll
      for (int v = 0; v < 8; ++v)
        qf[kh].w[v] = *(const u32*)(qrow + kh * 32 + a_kofs(v) + 16 * hf);
  }

  float mrow[8], lrow[8];
  AccF o[4];
#pragma unroll
  for (int r = 0; r < 8; ++r) { mrow[r] = -1e30f; lrow[r] = 0.f; }
#pragma unroll
  for (int t = 0; t < 4; ++t) { v8f z = {0,0,0,0,0,0,0,0}; o[t].v = z; }

  u16* pb = &Pbuf[wid][0];

  // per-wave TDM: stage first V tile (32 keys x 64 hd)
  tdm_load_2d((u32)(uintptr_t)&Vt[wid][0][0], vbase, 64, T_SEQ, 64, 32, (u64)C3);

  for (int kv0 = 0; kv0 < nkv; kv0 += 32) {
    const int cur = (kv0 >> 5) & 1;
    const bool more = (kv0 + 32) < nkv;
    if (more)                                           // prefetch next V tile
      tdm_load_2d((u32)(uintptr_t)&Vt[wid][cur ^ 1][0],
                  vbase + (size_t)(kv0 + 32) * C3, 64,
                  (u32)(T_SEQ - (kv0 + 32)), 64, 32, (u64)C3);
    if (more) __builtin_amdgcn_s_wait_tensorcnt(1);     // in-order: current landed
    else      __builtin_amdgcn_s_wait_tensorcnt(0);

    AccF s[2];
#pragma unroll
    for (int nt = 0; nt < 2; ++nt) {                    // S = Q * K^T
      const u16* krow = base + CED + (size_t)(kv0 + nt * 16 + lm) * C3;
      FragA kf0, kf1;
#pragma unroll
      for (int v = 0; v < 8; ++v) {
        kf0.w[v] = *(const u32*)(krow + 16 * hf + 2 * v);
        kf1.w[v] = *(const u32*)(krow + 32 + 16 * hf + 2 * v);
      }
      v8f z = {0,0,0,0,0,0,0,0};
      s[nt].v = __builtin_amdgcn_wmma_f32_16x16x32_bf16(
          false, qf[0].v, false, kf0.v, (short)0, z, false, false);
      s[nt].v = __builtin_amdgcn_wmma_f32_16x16x32_bf16(
          false, qf[1].v, false, kf1.v, (short)0, s[nt].v, false, false);
    }
    // scale + causal mask
#pragma unroll
    for (int nt = 0; nt < 2; ++nt) {
      int key = kv0 + nt * 16 + lm;
#pragma unroll
      for (int r = 0; r < 8; ++r) {
        int q = q0 + r + 8 * hf;
        float sv = s[nt].f[r] * 0.125f;                 // 1/sqrt(64)
        s[nt].f[r] = (key <= q) ? sv : -1e30f;
      }
    }
    // online softmax; row stats align with accumulator rows (m = r + 8*hf)
#pragma unroll
    for (int r = 0; r < 8; ++r) {
      float v = fmaxf(s[0].f[r], s[1].f[r]);
      v = fmaxf(v, __shfl_xor(v, 1, 16));
      v = fmaxf(v, __shfl_xor(v, 2, 16));
      v = fmaxf(v, __shfl_xor(v, 4, 16));
      v = fmaxf(v, __shfl_xor(v, 8, 16));
      float mnew = fmaxf(mrow[r], v);
      float alpha = __expf(mrow[r] - mnew);
      mrow[r] = mnew;
      lrow[r] *= alpha;
#pragma unroll
      for (int t = 0; t < 4; ++t) o[t].f[r] *= alpha;
    }
#pragma unroll
    for (int r = 0; r < 8; ++r) {
      float p0 = __expf(s[0].f[r] - mrow[r]);
      float p1 = __expf(s[1].f[r] - mrow[r]);
      s[0].f[r] = p0; s[1].f[r] = p1;
      float v = p0 + p1;
      v += __shfl_xor(v, 1, 16);
      v += __shfl_xor(v, 2, 16);
      v += __shfl_xor(v, 4, 16);
      v += __shfl_xor(v, 8, 16);
      lrow[r] += v;
    }
    // P: accumulator layout -> A-frag layout via per-wave LDS
#pragma unroll
    for (int nt = 0; nt < 2; ++nt)
#pragma unroll
      for (int r = 0; r < 8; ++r)
        pb[(r + 8 * hf) * 32 + nt * 16 + lm] = f2bf(s[nt].f[r]);
    asm volatile("s_wait_dscnt 0x0" ::: "memory");
    FragA pf;
    {
      const u16* prow = pb + lm * 32;
#pragma unroll
      for (int v = 0; v < 8; ++v) pf.w[v] = *(const u32*)(prow + a_kofs(v) + 16 * hf);
    }
    // O += P * V  (V from LDS tile: row k = kv0+16*hf+e, col = t*16+lm)
    const u16* vt = &Vt[wid][cur][0];
#pragma unroll
    for (int t = 0; t < 4; ++t) {
      FragA vf;
#pragma unroll
      for (int e = 0; e < 16; ++e)
        vf.u[e] = vt[(16 * hf + e) * 64 + t * 16 + lm];
      o[t].v = __builtin_amdgcn_wmma_f32_16x16x32_bf16(
          false, pf.v, false, vf.v, (short)0, o[t].v, false, false);
    }
    asm volatile("s_wait_dscnt 0x0" ::: "memory");      // WAR guard on Pbuf
  }
  // normalize and store
#pragma unroll
  for (int r = 0; r < 8; ++r) {
    float inv = 1.0f / lrow[r];
    int q = q0 + r + 8 * hf;
    u16* yrow = y + (size_t)(b * T_SEQ + q) * CED + h * 64;
#pragma unroll
    for (int t = 0; t < 4; ++t)
      yrow[t * 16 + lm] = f2bf(o[t].f[r] * inv);
  }
}

// ---------- orchestration ----------
extern "C" void kernel_launch(void* const* d_in, const int* in_sizes, int n_in,
                              void* d_out, int out_size, void* d_ws, size_t ws_size,
                              hipStream_t stream)
{
  (void)in_sizes; (void)n_in; (void)out_size; (void)ws_size;
  const float* x      = (const float*)d_in[0];
  const float* ln1w   = (const float*)d_in[1];
  const float* ln1b   = (const float*)d_in[2];
  const float* ln2w   = (const float*)d_in[3];
  const float* ln2b   = (const float*)d_in[4];
  const float* w_attn = (const float*)d_in[5];
  const float* b_attn = (const float*)d_in[6];
  const float* w_proj = (const float*)d_in[7];
  const float* b_proj = (const float*)d_in[8];
  const float* w_fc   = (const float*)d_in[9];
  const float* b_fc   = (const float*)d_in[10];
  const float* w_fc2  = (const float*)d_in[11];
  const float* b_fc2  = (const float*)d_in[12];

  char* ws = (char*)d_ws;
  const size_t MB = 1ull << 20;
  u16*   h    = (u16*)(ws + 0);           // 16 MB (reused for LN2 output)
  u16*   qkv  = (u16*)(ws + 16 * MB);     // 48 MB
  u16*   yb   = (u16*)(ws + 64 * MB);     // 16 MB
  float* x1   = (float*)(ws + 80 * MB);   // 32 MB
  u16*   mb   = (u16*)(ws + 112 * MB);    // 64 MB
  u16*   wat  = (u16*)(ws + 176 * MB);    // 6 MB   w_attn^T bf16 [3072,1024]
  u16*   wpt  = (u16*)(ws + 182 * MB);    // 2 MB   w_proj^T bf16 [1024,1024]
  u16*   wft  = (u16*)(ws + 184 * MB);    // 8 MB   w_fc^T   bf16 [4096,1024]
  u16*   wf2t = (u16*)(ws + 192 * MB);    // 8 MB   w_fc2^T  bf16 [1024,4096]

  const int BT = 4 * 2048;

  transpose_w_bf16<<<dim3(3072 / 32, 1024 / 32), 256, 0, stream>>>(w_attn, wat, 1024, 3072);
  transpose_w_bf16<<<dim3(1024 / 32, 1024 / 32), 256, 0, stream>>>(w_proj, wpt, 1024, 1024);
  transpose_w_bf16<<<dim3(4096 / 32, 1024 / 32), 256, 0, stream>>>(w_fc,   wft, 1024, 4096);
  transpose_w_bf16<<<dim3(1024 / 32, 4096 / 32), 256, 0, stream>>>(w_fc2, wf2t, 4096, 1024);

  layernorm_k<<<BT, 256, 0, stream>>>(x, ln1w, ln1b, h);
  gemm_wmma<0><<<dim3(3072 / BN, BT / BM), 256, 0, stream>>>(
      h, wat, b_attn, nullptr, qkv, BT, 3072, 1024);
  attn_flash<<<(4 * 16 * (2048 / 16)) / 4, 128, 0, stream>>>(qkv, yb);
  gemm_wmma<2><<<dim3(1024 / BN, BT / BM), 256, 0, stream>>>(
      yb, wpt, b_proj, x, x1, BT, 1024, 1024);
  layernorm_k<<<BT, 256, 0, stream>>>(x1, ln2w, ln2b, h);
  gemm_wmma<1><<<dim3(4096 / BN, BT / BM), 256, 0, stream>>>(
      h, wft, b_fc, nullptr, mb, BT, 4096, 1024);
  gemm_wmma<2><<<dim3(1024 / BN, BT / BM), 256, 0, stream>>>(
      mb, wf2t, b_fc2, x1, (float*)d_out, BT, 1024, 4096);
}
